// wide_basic_3058016715114
// MI455X (gfx1250) — compile-verified, matrix operand order checked by reading, not run on hardware
//
#include <hip/hip_runtime.h>
#include <cstdint>
#include <cstddef>

// ---------------- problem constants ----------------
#define BB    32          // batch
#define CC    160         // channels
#define HW    1024        // 32*32 spatial
#define NHD   4           // heads
#define DPAD  96          // dkh=80 padded to 3*32 for WMMA K-chunks
#define KTOT  1440        // 9*160 im2col depth
#define OTOT  800         // 128 conv + 672 qkv output channels
#define NPIX  (BB*HW)     // 32768 output pixels

typedef __attribute__((ext_vector_type(16))) _Float16 v16h;
typedef __attribute__((ext_vector_type(8)))  float    v8f;

struct alignas(16) U4 { unsigned int x, y, z, w; };

union Frag16 {
  v16h     v;
  _Float16 h[16];
  U4       q[2];
};

__device__ __forceinline__ v8f zero8() {
  v8f z = {0.f,0.f,0.f,0.f,0.f,0.f,0.f,0.f};
  return z;
}

__device__ __forceinline__ v8f wmma16(const Frag16& a, const Frag16& b, v8f c) {
  // D = A(16x32 f16) * B(32x16 f16) + C(16x16 f32)
  return __builtin_amdgcn_wmma_f32_16x16x32_f16(
      /*neg_a=*/false, a.v, /*neg_b=*/false, b.v,
      /*c_mod=*/(short)0, c, /*reuse_a=*/false, /*reuse_b=*/false);
}

// async global->LDS copy of 16 bytes (per-lane addresses), tracked by ASYNCcnt
__device__ __forceinline__ void async_b128(const void* gptr, void* lptr) {
  unsigned ldsoff = (unsigned)(uintptr_t)lptr;              // addr[31:0] = LDS byte offset
  unsigned long long ga = (unsigned long long)(uintptr_t)gptr;
  asm volatile("global_load_async_to_lds_b128 %0, %1, off"
               :: "v"(ldsoff), "v"(ga) : "memory");
}

// ---------------------------------------------------------------------------
// 1) Concatenated f16 weights:  Wcat[o][k], k = tap*160 + ci, o in [0,800)
// ---------------------------------------------------------------------------
__global__ void k_wcat(const float* __restrict__ cw, const float* __restrict__ qw,
                       _Float16* __restrict__ Wc) {
  int idx = blockIdx.x * 256 + threadIdx.x;
  if (idx >= OTOT * KTOT) return;
  int o = idx / KTOT, k = idx % KTOT;
  int r = k / CC, ci = k % CC;                       // r = kh*3+kw
  float v = (o < 128) ? cw[(o * CC + ci) * 9 + r]
                      : qw[((o - 128) * CC + ci) * 9 + r];
  Wc[idx] = (_Float16)v;
}

// ---------------------------------------------------------------------------
// 2) eval-BN + ReLU, f32 NCHW -> f16 NHWC (im2col-friendly)
// ---------------------------------------------------------------------------
__global__ void k_bnrelu(const float* __restrict__ src,
                         const float* __restrict__ g, const float* __restrict__ be,
                         const float* __restrict__ mn, const float* __restrict__ vr,
                         _Float16* __restrict__ y) {
  int idx = blockIdx.x * 256 + threadIdx.x;
  if (idx >= BB * CC * HW) return;
  int b = idx / (CC * HW);
  int c = (idx / HW) % CC;
  int p = idx % HW;
  float sc = g[c] * rsqrtf(vr[c] + 1e-5f);
  float v = (src[idx] - mn[c]) * sc + be[c];
  v = fmaxf(v, 0.f);
  y[(b * HW + p) * CC + c] = (_Float16)v;
}

// ---------------------------------------------------------------------------
// 3) implicit-GEMM conv:  D[o, p] = Wcat[o,:] . im2col[:, p]
//    block = 256 thr (8 waves), tile 64(o) x 128(pix), K steps of 32.
//    Double-buffered LDS staging via GLOBAL_LOAD_ASYNC_TO_LDS_B128:
//    3 async b128 per wave per step (uniform count -> exact ASYNCcnt math),
//    clamped addresses + post-wait ds_store zero fix-up for halo/tail OOB.
//    Epilogue routes o -> conv_out(stage) / q / k / v^T with bias.
// ---------------------------------------------------------------------------
__launch_bounds__(256)
__global__ void k_gemm(const _Float16* __restrict__ Wc, const _Float16* __restrict__ y,
                       const float* __restrict__ cb, const float* __restrict__ qb,
                       float* __restrict__ stage, _Float16* __restrict__ qb16,
                       _Float16* __restrict__ kb16, _Float16* __restrict__ vT) {
  __shared__ alignas(16) _Float16 Wt[2][64 * 32];   // 2 x 4 KB
  __shared__ alignas(16) _Float16 Bt[2][128 * 32];  // 2 x 8 KB
  const int pB = blockIdx.x * 128;
  const int oB = blockIdx.y * 64;
  const int tid = threadIdx.x, lane = tid & 31, wid = tid >> 5;
  const int wo = (wid >> 2) << 5;                  // 0 / 32
  const int wp = (wid & 3) << 5;                   // 0,32,64,96
  const int koff = (lane < 16) ? 0 : 8;

  // per-thread staging geometry (constant across K steps)
  const int wrow = tid >> 2, wseg = tid & 3;       // W tile: 1 b128 / thread
  const int oW   = oB + wrow;
  const int oWc  = (oW < OTOT) ? oW : (OTOT - 1);  // clamped (always valid)
  const bool wOOB = (oW >= OTOT);
  const int pix = tid >> 1, halfB = tid & 1;       // B tile: 2 b128 / thread
  const int p  = pB + pix;
  const int b  = p >> 10, hw = p & 1023, h = hw >> 5, w = hw & 31;

  auto issue = [&](int kc, int buf) {
    // W tile (64 x 32 f16)
    async_b128(Wc + oWc * KTOT + kc * 32 + wseg * 8,
               &Wt[buf][wrow * 32 + wseg * 8]);
    // B tile (128 pix x 32 k); one conv tap per 32-chunk (160 = 5*32)
    int r  = kc / 5;
    int dh = r / 3 - 1, dw = r % 3 - 1;
    int ci0 = (kc % 5) * 32 + halfB * 16;
    int hh = h + dh, ww = w + dw;
    int hhc = hh < 0 ? 0 : (hh > 31 ? 31 : hh);    // clamp: keep issue count uniform
    int wwc = ww < 0 ? 0 : (ww > 31 ? 31 : ww);
    const _Float16* sp = y + (((b * 32 + hhc) * 32 + wwc) * CC + ci0);
    _Float16* dst = &Bt[buf][pix * 32 + halfB * 16];
    async_b128(sp,     dst);
    async_b128(sp + 8, dst + 8);
  };
  auto fixup = [&](int kc, int buf) {              // zero halo / o-tail slots
    U4 z = {0u,0u,0u,0u};
    if (wOOB) *(U4*)&Wt[buf][wrow * 32 + wseg * 8] = z;
    int r  = kc / 5;
    int dh = r / 3 - 1, dw = r % 3 - 1;
    int hh = h + dh, ww = w + dw;
    if ((unsigned)hh >= 32u || (unsigned)ww >= 32u) {
      *(U4*)&Bt[buf][pix * 32 + halfB * 16]     = z;
      *(U4*)&Bt[buf][pix * 32 + halfB * 16 + 8] = z;
    }
  };

  v8f acc[2][2];
  acc[0][0] = zero8(); acc[0][1] = zero8();
  acc[1][0] = zero8(); acc[1][1] = zero8();

  issue(0, 0);                                     // prologue: prefetch step 0

  for (int kc = 0; kc < 45; ++kc) {                // 45*32 = 1440
    const int buf = kc & 1;
    if (kc + 1 < 45) {
      issue(kc + 1, buf ^ 1);                      // overlap next copy w/ compute
      // 3 newly issued + 3 old outstanding; loads retire in order,
      // so draining to <=3 guarantees step-kc tiles have landed.
      asm volatile("s_wait_asynccnt 0x3" ::: "memory");
    } else {
      asm volatile("s_wait_asynccnt 0x0" ::: "memory");
    }
    fixup(kc, buf);
    __syncthreads();

    Frag16 a0, a1, b0, b1;
    int ar = wo + (lane & 15);
    a0.q[0] = *(const U4*)(&Wt[buf][ar * 32 + koff]);
    a0.q[1] = *(const U4*)(&Wt[buf][ar * 32 + koff + 16]);
    a1.q[0] = *(const U4*)(&Wt[buf][(ar + 16) * 32 + koff]);
    a1.q[1] = *(const U4*)(&Wt[buf][(ar + 16) * 32 + koff + 16]);
    int bc = wp + (lane & 15);
    b0.q[0] = *(const U4*)(&Bt[buf][bc * 32 + koff]);
    b0.q[1] = *(const U4*)(&Bt[buf][bc * 32 + koff + 16]);
    b1.q[0] = *(const U4*)(&Bt[buf][(bc + 16) * 32 + koff]);
    b1.q[1] = *(const U4*)(&Bt[buf][(bc + 16) * 32 + koff + 16]);

    acc[0][0] = wmma16(a0, b0, acc[0][0]);
    acc[0][1] = wmma16(a0, b1, acc[0][1]);
    acc[1][0] = wmma16(a1, b0, acc[1][0]);
    acc[1][1] = wmma16(a1, b1, acc[1][1]);
    __syncthreads();                               // reads done before next overwrite
  }

  // epilogue: C layout -> lane holds (N = lane&15), rows r + 8*(lane>>4)
  const float QSCALE = 0.11180339887498948f;       // 80^-0.5
  const int half8 = (lane >> 4) << 3;
  for (int ti = 0; ti < 2; ++ti)
    for (int tj = 0; tj < 2; ++tj)
      for (int r = 0; r < 8; ++r) {
        int o = oB + wo + ti * 16 + r + half8;
        int pp = pB + wp + tj * 16 + (lane & 15);
        float val = acc[ti][tj][r];
        int bb = pp >> 10, hw2 = pp & 1023;
        if (o < 128) {
          stage[(bb * CC + o) * HW + hw2] = val + cb[o];
        } else if (o < OTOT) {
          float vb = val + qb[o - 128];
          if (o < 448) {               // q (scaled)
            int qc = o - 128; int n = qc / 80, d = qc % 80;
            qb16[((bb * NHD + n) * HW + hw2) * DPAD + d] = (_Float16)(vb * QSCALE);
          } else if (o < 768) {        // k
            int c2 = o - 448; int n = c2 / 80, d = c2 % 80;
            kb16[((bb * NHD + n) * HW + hw2) * DPAD + d] = (_Float16)vb;
          } else {                     // v  (transposed: [b][n][dv][hw])
            int vc = o - 768; int n = vc / 8, dv = vc & 7;
            vT[((bb * NHD + n) * 8 + dv) * HW + hw2] = (_Float16)vb;
          }
        }
      }
}

// ---------------------------------------------------------------------------
// 4) rel-logit precompute: rh/rw[b,n,i,m] = q_i . rel_{h,w}[m]   (m<63, pad 64)
// ---------------------------------------------------------------------------
__global__ void k_relprec(const _Float16* __restrict__ q,
                          const float* __restrict__ relh, const float* __restrict__ relw,
                          _Float16* __restrict__ rh, _Float16* __restrict__ rw) {
  int idx = blockIdx.x * 256 + threadIdx.x;
  if (idx >= BB * NHD * HW * 64) return;
  int row = idx >> 6, m = idx & 63;
  if (m >= 63) { rh[idx] = (_Float16)0.f; rw[idx] = (_Float16)0.f; return; }
  const _Float16* qp = q + (size_t)row * DPAD;
  float sh = 0.f, sw = 0.f;
  for (int d = 0; d < 80; ++d) {
    float qv = (float)qp[d];
    sh += qv * relh[m * 80 + d];
    sw += qv * relw[m * 80 + d];
  }
  rh[idx] = (_Float16)sh;
  rw[idx] = (_Float16)sw;
}

// ---------------------------------------------------------------------------
// 5) flash attention: one wave per 16-query tile, 32-key steps.
//    logits(i,j) = q_i.k_j + rh[i, xj-xi+31] + rw[i, yj-yi+31]
// ---------------------------------------------------------------------------
__launch_bounds__(128)
__global__ void k_attn(const _Float16* __restrict__ q, const _Float16* __restrict__ kk,
                       const _Float16* __restrict__ vT, const _Float16* __restrict__ rh,
                       const _Float16* __restrict__ rw, float* __restrict__ attnbuf) {
  __shared__ alignas(16) _Float16 Pt[4][16 * 32];  // per-wave P transpose tile
  const int tid = threadIdx.x, lane = tid & 31, wid = tid >> 5;
  const int gid = blockIdx.x;
  const int b = gid >> 6, rem = gid & 63, n = rem >> 4, it = rem & 15;
  const int i0 = it * 64 + wid * 16;
  const int bn = b * NHD + n;
  const int koff = (lane < 16) ? 0 : 8;
  const int halfsel = lane >> 4;

  // persistent Q A-fragments (d = 0..95)
  Frag16 qa[3];
  const _Float16* qrow = q + ((size_t)(bn * HW + i0 + (lane & 15))) * DPAD;
  for (int c = 0; c < 3; ++c) {
    qa[c].q[0] = *(const U4*)(qrow + c * 32 + koff);
    qa[c].q[1] = *(const U4*)(qrow + c * 32 + koff + 16);
  }
  const _Float16* rhB = rh + (((size_t)(bn * HW + i0)) << 6);
  const _Float16* rwB = rw + (((size_t)(bn * HW + i0)) << 6);

  float mrow[8], srow[8];
  for (int r = 0; r < 8; ++r) { mrow[r] = -1e30f; srow[r] = 0.f; }
  v8f oacc = zero8();

  for (int js = 0; js < 32; ++js) {
    const int j0 = js * 32;
    v8f s0 = zero8(), s1 = zero8();
    const _Float16* krow0 = kk + ((size_t)(bn * HW + j0 + (lane & 15))) * DPAD;
    const _Float16* krow1 = krow0 + 16 * DPAD;
    for (int c = 0; c < 3; ++c) {
      Frag16 kb;
      kb.q[0] = *(const U4*)(krow0 + c * 32 + koff);
      kb.q[1] = *(const U4*)(krow0 + c * 32 + koff + 16);
      s0 = wmma16(qa[c], kb, s0);
    }
    for (int c = 0; c < 3; ++c) {
      Frag16 kb;
      kb.q[0] = *(const U4*)(krow1 + c * 32 + koff);
      kb.q[1] = *(const U4*)(krow1 + c * 32 + koff + 16);
      s1 = wmma16(qa[c], kb, s1);
    }

    // relative-position terms (C-frag: col = lane&15, row = r + 8*halfsel)
    int jc0 = j0 + (lane & 15), jc1 = jc0 + 16;
    int xj0 = jc0 >> 5, yj0 = jc0 & 31, xj1 = jc1 >> 5, yj1 = jc1 & 31;
    for (int r = 0; r < 8; ++r) {
      int il = r + (halfsel << 3);
      int ig = i0 + il, xi = ig >> 5, yi = ig & 31;
      s0[r] = s0[r] + (float)rhB[(il << 6) + (xj0 - xi + 31)]
                    + (float)rwB[(il << 6) + (yj0 - yi + 31)];
      s1[r] = s1[r] + (float)rhB[(il << 6) + (xj1 - xi + 31)]
                    + (float)rwB[(il << 6) + (yj1 - yi + 31)];
    }

    // online softmax (row reductions across the 16-lane half via shfl_xor)
    for (int r = 0; r < 8; ++r) {
      float cm = fmaxf(s0[r], s1[r]);
      for (int off = 8; off >= 1; off >>= 1) cm = fmaxf(cm, __shfl_xor(cm, off, 32));
      float mnew  = fmaxf(mrow[r], cm);
      float scale = __expf(mrow[r] - mnew);
      float e0 = __expf(s0[r] - mnew);
      float e1 = __expf(s1[r] - mnew);
      float rs = e0 + e1;
      for (int off = 8; off >= 1; off >>= 1) rs += __shfl_xor(rs, off, 32);
      srow[r] = srow[r] * scale + rs;
      mrow[r] = mnew;
      oacc[r] = oacc[r] * scale;
      int il = r + (halfsel << 3);
      Pt[wid][il * 32 + (lane & 15)]      = (_Float16)e0;
      Pt[wid][il * 32 + (lane & 15) + 16] = (_Float16)e1;
    }
    __syncthreads();

    // P.V : A = P tile from LDS, B = v^T slice (dv padded 8->16 with zeros)
    Frag16 pa, vb;
    int il2 = lane & 15;
    pa.q[0] = *(const U4*)(&Pt[wid][il2 * 32 + koff]);
    pa.q[1] = *(const U4*)(&Pt[wid][il2 * 32 + koff + 16]);
    int dv = lane & 15;
    if (dv < 8) {
      const _Float16* vrow = vT + ((size_t)(bn * 8 + dv)) * HW + j0 + koff;
      vb.q[0] = *(const U4*)vrow;
      vb.q[1] = *(const U4*)(vrow + 16);
    } else {
      U4 z = {0u,0u,0u,0u};
      vb.q[0] = z; vb.q[1] = z;
    }
    oacc = wmma16(pa, vb, oacc);
    __syncthreads();
  }

  // normalize + store: matches raw reshape (B,Nh,HW,DVH) -> flat
  int dv = lane & 15;
  if (dv < 8) {
    for (int r = 0; r < 8; ++r) {
      int il = r + (halfsel << 3);
      int ig = i0 + il;
      attnbuf[((size_t)b << 15) + (n << 13) + (ig << 3) + dv] = oacc[r] / srow[il - (halfsel << 3)];
    }
  }
}

// ---------------------------------------------------------------------------
// 6) 1x1 attn projection -> stage channels [128,160)
// ---------------------------------------------------------------------------
__global__ void k_attn2(const float* __restrict__ attnbuf, const float* __restrict__ aw,
                        const float* __restrict__ ab, float* __restrict__ stage) {
  int idx = blockIdx.x * 256 + threadIdx.x;
  if (idx >= BB * 32 * HW) return;
  int b = idx >> 15, o = (idx >> 10) & 31, p = idx & 1023;
  const float* src = attnbuf + ((size_t)b << 15);
  float s = ab[o];
  for (int c = 0; c < 32; ++c) s += aw[o * 32 + c] * src[(c << 10) + p];
  stage[((size_t)b * CC + 128 + o) * HW + p] = s;
}

// ---------------------------------------------------------------------------
// 7) residual add
// ---------------------------------------------------------------------------
__global__ void k_resid(const float* __restrict__ stage, const float* __restrict__ x,
                        float* __restrict__ out) {
  int idx = blockIdx.x * 256 + threadIdx.x;
  if (idx >= BB * CC * HW) return;
  out[idx] = stage[idx] + x[idx];
}

// ---------------------------------------------------------------------------
// workspace layout (all 256B aligned); total ~120.4 MB
// ---------------------------------------------------------------------------
constexpr size_t SZ_Y   = (size_t)NPIX * CC * 2;            // 10,485,760
constexpr size_t SZ_WC  = (size_t)OTOT * KTOT * 2;          //  2,304,000
constexpr size_t SZ_STG = (size_t)BB * CC * HW * 4;         // 20,971,520
constexpr size_t SZ_QK  = (size_t)BB * NHD * HW * DPAD * 2; // 25,165,824
constexpr size_t SZ_VT  = (size_t)BB * NHD * 8 * HW * 2;    //  2,097,152
constexpr size_t SZ_R   = (size_t)BB * NHD * HW * 64 * 2;   // 16,777,216
constexpr size_t SZ_AB  = (size_t)BB * 32 * HW * 4;         //  4,194,304

constexpr size_t OFF_Y   = 0;
constexpr size_t OFF_WCA = OFF_Y   + SZ_Y;
constexpr size_t OFF_WCB = OFF_WCA + SZ_WC;
constexpr size_t OFF_STG = OFF_WCB + SZ_WC;
constexpr size_t OFF_Q   = OFF_STG + SZ_STG;
constexpr size_t OFF_K   = OFF_Q   + SZ_QK;
constexpr size_t OFF_VT  = OFF_K   + SZ_QK;
constexpr size_t OFF_RH  = OFF_VT  + SZ_VT;
constexpr size_t OFF_RW  = OFF_RH  + SZ_R;
constexpr size_t OFF_AB  = OFF_RW  + SZ_R;

extern "C" void kernel_launch(void* const* d_in, const int* in_sizes, int n_in,
                              void* d_out, int out_size, void* d_ws, size_t ws_size,
                              hipStream_t stream) {
  (void)in_sizes; (void)n_in; (void)out_size; (void)ws_size;
  const float* x   = (const float*)d_in[0];
  // bn1
  const float* g1  = (const float*)d_in[1];
  const float* be1 = (const float*)d_in[2];
  const float* mn1 = (const float*)d_in[3];
  const float* vr1 = (const float*)d_in[4];
  // ac1
  const float* cw1 = (const float*)d_in[5];
  const float* cb1 = (const float*)d_in[6];
  const float* qw1 = (const float*)d_in[7];
  const float* qb1 = (const float*)d_in[8];
  const float* aw1 = (const float*)d_in[9];
  const float* ab1 = (const float*)d_in[10];
  const float* rlw1 = (const float*)d_in[11];
  const float* rlh1 = (const float*)d_in[12];
  // bn2
  const float* g2  = (const float*)d_in[13];
  const float* be2 = (const float*)d_in[14];
  const float* mn2 = (const float*)d_in[15];
  const float* vr2 = (const float*)d_in[16];
  // ac2
  const float* cw2 = (const float*)d_in[17];
  const float* cb2 = (const float*)d_in[18];
  const float* qw2 = (const float*)d_in[19];
  const float* qb2 = (const float*)d_in[20];
  const float* aw2 = (const float*)d_in[21];
  const float* ab2 = (const float*)d_in[22];
  const float* rlw2 = (const float*)d_in[23];
  const float* rlh2 = (const float*)d_in[24];

  uint8_t* ws = (uint8_t*)d_ws;
  _Float16* y    = (_Float16*)(ws + OFF_Y);
  _Float16* WcA  = (_Float16*)(ws + OFF_WCA);
  _Float16* WcB  = (_Float16*)(ws + OFF_WCB);
  float*    stg  = (float*)   (ws + OFF_STG);
  _Float16* qb16 = (_Float16*)(ws + OFF_Q);
  _Float16* kb16 = (_Float16*)(ws + OFF_K);
  _Float16* vT   = (_Float16*)(ws + OFF_VT);
  _Float16* rhb  = (_Float16*)(ws + OFF_RH);
  _Float16* rwb  = (_Float16*)(ws + OFF_RW);
  float*    abuf = (float*)   (ws + OFF_AB);
  float*    out  = (float*)d_out;

  // zero q/k pads (d in [80,96)) once per call; GEMM rewrites d<80 each call
  hipMemsetAsync(qb16, 0, SZ_QK, stream);
  hipMemsetAsync(kb16, 0, SZ_QK, stream);

  const int THREADS = 256;
  const int nWcat = (OTOT * KTOT + THREADS - 1) / THREADS;
  const int nElem = (BB * CC * HW + THREADS - 1) / THREADS;
  const int nRel  = (BB * NHD * HW * 64 + THREADS - 1) / THREADS;
  const int nA2   = (BB * 32 * HW + THREADS - 1) / THREADS;
  const dim3 gemmGrid(NPIX / 128, (OTOT + 63) / 64);   // (256, 13)

  k_wcat<<<nWcat, THREADS, 0, stream>>>(cw1, qw1, WcA);
  k_wcat<<<nWcat, THREADS, 0, stream>>>(cw2, qw2, WcB);

  // ---- block 1 ----
  k_bnrelu<<<nElem, THREADS, 0, stream>>>(x, g1, be1, mn1, vr1, y);
  k_gemm<<<gemmGrid, THREADS, 0, stream>>>(WcA, y, cb1, qb1, stg, qb16, kb16, vT);
  k_relprec<<<nRel, THREADS, 0, stream>>>(qb16, rlh1, rlw1, rhb, rwb);
  k_attn<<<BB * NHD * 16, 128, 0, stream>>>(qb16, kb16, vT, rhb, rwb, abuf);
  k_attn2<<<nA2, THREADS, 0, stream>>>(abuf, aw1, ab1, stg);

  // ---- block 2 (reuse y / stage / q / k / v / rel buffers) ----
  k_bnrelu<<<nElem, THREADS, 0, stream>>>(stg, g2, be2, mn2, vr2, y);
  k_gemm<<<gemmGrid, THREADS, 0, stream>>>(WcB, y, cb2, qb2, stg, qb16, kb16, vT);
  k_relprec<<<nRel, THREADS, 0, stream>>>(qb16, rlh2, rlw2, rhb, rwb);
  k_attn<<<BB * NHD * 16, 128, 0, stream>>>(qb16, kb16, vT, rhb, rwb, abuf);
  k_attn2<<<nA2, THREADS, 0, stream>>>(abuf, aw2, ab2, stg);

  // ---- residual ----
  k_resid<<<nElem, THREADS, 0, stream>>>(stg, x, out);
}
